// DeformConvOriginalDepthWise_20109036880559
// MI455X (gfx1250) — matrix operand order repaired and checked by reading, compiled, not run on hardware
//
#include <hip/hip_runtime.h>

// ---- problem constants (from reference) ----
#define Hd    96
#define Wd    96
#define HWD   (Hd * Wd)        // 9216
#define CIN   256
#define COUT  256
#define BATCH 2
#define KK    9                // 3x3 taps
#define NOFF  18               // 2*K*K offset channels

typedef __bf16 bf16_t;
typedef __attribute__((ext_vector_type(16))) __bf16 v16bf;
typedef __attribute__((ext_vector_type(8)))  __bf16 v8bf;
typedef __attribute__((ext_vector_type(8)))  float  v8f;
typedef __attribute__((ext_vector_type(4)))  int    v4i;

union V16 { v16bf v; v8bf h[2]; };

// ---- CDNA5 async global->LDS path (guarded: falls back to plain copies) ----
#if defined(__has_builtin)
#if __has_builtin(__builtin_amdgcn_global_load_async_to_lds_b128)
#define HAVE_ASYNC_LDS 1
#endif
#endif

#ifdef HAVE_ASYNC_LDS
typedef __attribute__((address_space(1))) v4i gv4i;   // global
typedef __attribute__((address_space(3))) v4i lv4i;   // LDS
// generic->AS1: numerically identical 64-bit address.
__device__ __forceinline__ gv4i* to_gptr(const void* p) {
    return (gv4i*)(unsigned long long)(p);
}
// generic->AS3: LDS aperture lives in addr[63:32]; low 32 bits are the LDS offset.
__device__ __forceinline__ lv4i* to_lptr(void* p) {
    return (lv4i*)(unsigned int)(unsigned long long)(p);
}
#if __has_builtin(__builtin_amdgcn_s_wait_asynccnt)
#define ASYNC_WAIT() __builtin_amdgcn_s_wait_asynccnt(0)
#else
#define ASYNC_WAIT() asm volatile("s_wait_asynccnt 0x0" ::: "memory")
#endif
#endif

// ---------------------------------------------------------------------------
// Kernel 0a: convert pw_w (256x256) and off_w (18x256 -> zero-padded 32x256)
// to bf16 in workspace.
// ---------------------------------------------------------------------------
__global__ __launch_bounds__(256) void convert_weights_kernel(
    const float* __restrict__ pw_w, const float* __restrict__ off_w,
    bf16_t* __restrict__ pwbf, bf16_t* __restrict__ owbf)
{
    int i = blockIdx.x * 256 + threadIdx.x;           // 0 .. 65535
    if (i < COUT * CIN) pwbf[i] = (bf16_t)pw_w[i];
    if (i < 32 * CIN) {
        int row = i >> 8;
        int col = i & 255;
        owbf[i] = (row < NOFF) ? (bf16_t)off_w[row * CIN + col] : (bf16_t)0.f;
    }
}

// ---------------------------------------------------------------------------
// Kernel 0b: transpose-convert x[b][c][hw] (fp32) -> xT[b][hw][c] (bf16).
// 32x32 tiles through LDS; both global sides coalesced.
// grid: (HWD/32, CIN/32, BATCH), block 256
// ---------------------------------------------------------------------------
__global__ __launch_bounds__(256) void transpose_x_kernel(
    const float* __restrict__ x, bf16_t* __restrict__ xT)
{
    __shared__ float tile[32][33];
    const int b   = blockIdx.z;
    const int c0  = blockIdx.y * 32;
    const int hw0 = blockIdx.x * 32;
    const int tj  = threadIdx.x & 31;
    const int ti  = threadIdx.x >> 5;  // 0..7
    const float* src = x + (size_t)b * CIN * HWD;
    bf16_t* dst = xT + (size_t)b * HWD * CIN;
#pragma unroll
    for (int r = 0; r < 4; r++) {
        const int row = ti + r * 8;    // c index in tile
        tile[row][tj] = src[(size_t)(c0 + row) * HWD + hw0 + tj];
    }
    __syncthreads();
#pragma unroll
    for (int r = 0; r < 4; r++) {
        const int row = ti + r * 8;    // hw index in tile
        dst[(size_t)(hw0 + row) * CIN + c0 + tj] = (bf16_t)tile[tj][row];
    }
}

// ---------------------------------------------------------------------------
// Shared GEMM core: block (128 thr = 4 waves) computes a 32(M) x 32(N) tile
// of D[m][n] = sum_k A[m][k]*B[k][n], K=256, via v_wmma_f32_16x16x32_bf16.
//   A : bf16 [M][256] row-major (K contiguous)
//   BT: bf16 [N][256] row-major (K contiguous)  == B transposed
// Each thread stages exactly one 16-byte chunk per tile per k-step, which on
// CDNA5 maps to GLOBAL_LOAD_ASYNC_TO_LDS_B128 (ASYNCcnt) when available.
// LDS row stride 40 halfs keeps 16B alignment and de-conflicts banks.
// Fragment reads match the ISA 16-bit A 16x32 / B 32x16 VGPR layouts.
// ---------------------------------------------------------------------------
#define LDS_STRIDE 40

__device__ __forceinline__ v8f gemm32x32_core(
    const bf16_t* __restrict__ A, int o0,
    const bf16_t* __restrict__ BT, int hw0,
    bf16_t* Als, bf16_t* Bls, int tid)
{
    const int lane = tid & 31;
    const int wv   = tid >> 5;            // 0..3
    const int mt   = (wv >> 1) * 16;
    const int nt   = (wv & 1) * 16;

    const int row = tid >> 2;             // 0..31
    const int col = (tid & 3) << 3;       // 0,8,16,24
    const bf16_t* ga = A  + (size_t)(o0  + row) * CIN + col;
    const bf16_t* gb = BT + (size_t)(hw0 + row) * CIN + col;
    bf16_t* la = Als + row * LDS_STRIDE + col;
    bf16_t* lb = Bls + row * LDS_STRIDE + col;

    const int ar = (mt + (lane & 15)) * LDS_STRIDE + ((lane >> 4) << 3);
    const int br = (nt + (lane & 15)) * LDS_STRIDE + ((lane >> 4) << 3);

    v8f acc = {};

    for (int k0 = 0; k0 < CIN; k0 += 32) {
#ifdef HAVE_ASYNC_LDS
        __builtin_amdgcn_global_load_async_to_lds_b128(to_gptr(ga + k0), to_lptr(la), 0, 0);
        __builtin_amdgcn_global_load_async_to_lds_b128(to_gptr(gb + k0), to_lptr(lb), 0, 0);
        ASYNC_WAIT();
#else
        *(v8bf*)la = *(const v8bf*)(ga + k0);
        *(v8bf*)lb = *(const v8bf*)(gb + k0);
#endif
        __syncthreads();

        V16 a, b;
        a.h[0] = *(const v8bf*)(Als + ar);
        a.h[1] = *(const v8bf*)(Als + ar + 16);
        b.h[0] = *(const v8bf*)(Bls + br);
        b.h[1] = *(const v8bf*)(Bls + br + 16);

        acc = __builtin_amdgcn_wmma_f32_16x16x32_bf16(
                  false, a.v, false, b.v, (short)0, acc, false, false);

        __syncthreads();
    }
    return acc;
}

// ---------------------------------------------------------------------------
// Kernel 1: xp[b][o][hw] = sum_c pw[o][c]*x[b][c][hw].
// Stores fp32 xp (channel-plane layout for the gather stage) AND a bf16
// transposed copy xpT[b][hw][o] (one contiguous 16B store per lane) for the
// offsets GEMM. grid: (HWD/32, COUT/32, BATCH), block 128
// ---------------------------------------------------------------------------
__global__ __launch_bounds__(128) void gemm_xp_kernel(
    const bf16_t* __restrict__ xT, const bf16_t* __restrict__ pwbf,
    float* __restrict__ xp, bf16_t* __restrict__ xpT)
{
    __shared__ bf16_t Als[32 * LDS_STRIDE];
    __shared__ bf16_t Bls[32 * LDS_STRIDE];

    const int b   = blockIdx.z;
    const int o0  = blockIdx.y * 32;
    const int hw0 = blockIdx.x * 32;
    const int tid = threadIdx.x;

    const bf16_t* BT = xT + (size_t)b * HWD * CIN;
    v8f acc = gemm32x32_core(pwbf, o0, BT, hw0, Als, Bls, tid);

    const int lane = tid & 31;
    const int wv   = tid >> 5;
    const int mt   = (wv >> 1) * 16;
    const int nt   = (wv & 1) * 16;
    const int n_g  = hw0 + nt + (lane & 15);
    const int m0   = o0 + mt + ((lane >> 4) << 3);

    v8bf pk;
#pragma unroll
    for (int i = 0; i < 8; i++) {
        xp[(size_t)(b * COUT + m0 + i) * HWD + n_g] = acc[i];
        pk[i] = (bf16_t)acc[i];
    }
    *(v8bf*)(xpT + ((size_t)b * HWD + n_g) * CIN + m0) = pk;
}

// ---------------------------------------------------------------------------
// Kernel 2: off[b][o][hw] = sum_c off_w[o][c]*xp[b][c][hw] + off_b[o].
// A zero-padded to 32 rows; B from bf16 xpT. grid: (HWD/32, 1, BATCH), 128thr
// ---------------------------------------------------------------------------
__global__ __launch_bounds__(128) void gemm_off_kernel(
    const bf16_t* __restrict__ xpT, const bf16_t* __restrict__ owbf,
    const float* __restrict__ off_b, float* __restrict__ off)
{
    __shared__ bf16_t Als[32 * LDS_STRIDE];
    __shared__ bf16_t Bls[32 * LDS_STRIDE];

    const int b   = blockIdx.z;
    const int hw0 = blockIdx.x * 32;
    const int tid = threadIdx.x;

    const bf16_t* BT = xpT + (size_t)b * HWD * CIN;
    v8f acc = gemm32x32_core(owbf, 0, BT, hw0, Als, Bls, tid);

    const int lane = tid & 31;
    const int wv   = tid >> 5;
    const int mt   = (wv >> 1) * 16;
    const int nt   = (wv & 1) * 16;
    const int n_g  = hw0 + nt + (lane & 15);
    const int m0   = mt + ((lane >> 4) << 3);
#pragma unroll
    for (int i = 0; i < 8; i++) {
        const int m_g = m0 + i;
        if (m_g < NOFF)
            off[(size_t)(b * NOFF + m_g) * HWD + n_g] = acc[i] + off_b[m_g];
    }
}

// ---------------------------------------------------------------------------
// Kernel 3: bilinear deform-sample + depthwise 3x3 combine.
// Block = 64 hw positions x 64 channels (256 thr). Phase 1 builds the 9 taps'
// 4 clamped indices + validity-masked weights per hw position (shared across
// all channels) in LDS. Phase 2 gathers xp (L2-resident) and accumulates.
// grid: (HWD/64, COUT/64, BATCH)
// ---------------------------------------------------------------------------
__global__ __launch_bounds__(256) void deform_sample_kernel(
    const float* __restrict__ xp, const float* __restrict__ off,
    const float* __restrict__ dw_w, float* __restrict__ out)
{
    __shared__ int   s_idx[KK * 64 * 4];
    __shared__ float s_wt [KK * 64 * 4];

    const int b   = blockIdx.z;
    const int c0  = blockIdx.y * 64;
    const int hw0 = blockIdx.x * 64;
    const int tid = threadIdx.x;

    const float* offb = off + (size_t)b * NOFF * HWD;

    for (int e = tid; e < KK * 64; e += 256) {
        const int tap = e >> 6;                 // 0..8
        const int pos = e & 63;
        const int hw  = hw0 + pos;
        const int h   = hw / Wd;
        const int w   = hw - h * Wd;
        const int ky  = tap / 3;
        const int kx  = tap - ky * 3;
        const float oy = offb[(size_t)(2 * tap    ) * HWD + hw];
        const float ox = offb[(size_t)(2 * tap + 1) * HWD + hw];
        const float y  = (float)(h - 1 + ky) + oy;
        const float x  = (float)(w - 1 + kx) + ox;
        const float y0 = floorf(y);
        const float x0 = floorf(x);
        const float ly = y - y0, lx = x - x0;
        float wts[4];
        wts[0] = (1.f - ly) * (1.f - lx);
        wts[1] = (1.f - ly) * lx;
        wts[2] = ly * (1.f - lx);
        wts[3] = ly * lx;
        const int base = e * 4;
#pragma unroll
        for (int j = 0; j < 4; j++) {
            const float yi = y0 + (float)(j >> 1);
            const float xi = x0 + (float)(j & 1);
            const bool valid = (yi >= 0.f) && (yi <= (float)(Hd - 1)) &&
                               (xi >= 0.f) && (xi <= (float)(Wd - 1));
            int yc = (int)yi; yc = yc < 0 ? 0 : (yc > Hd - 1 ? Hd - 1 : yc);
            int xc = (int)xi; xc = xc < 0 ? 0 : (xc > Wd - 1 ? Wd - 1 : xc);
            s_idx[base + j] = yc * Wd + xc;
            s_wt [base + j] = valid ? wts[j] : 0.f;
        }
    }
    __syncthreads();

    const int pos = tid & 63;
    const int hw  = hw0 + pos;
    const int cl0 = tid >> 6;                   // 0..3

    for (int pass = 0; pass < 16; pass++) {
        const int c = c0 + cl0 + pass * 4;
        const float* chan = xp + (size_t)(b * COUT + c) * HWD;
        const float* dw   = dw_w + c * KK;
        float acc = 0.f;
#pragma unroll
        for (int tap = 0; tap < KK; tap++) {
            const int base = (tap * 64 + pos) * 4;
            float s = s_wt[base + 0] * chan[s_idx[base + 0]]
                    + s_wt[base + 1] * chan[s_idx[base + 1]]
                    + s_wt[base + 2] * chan[s_idx[base + 2]]
                    + s_wt[base + 3] * chan[s_idx[base + 3]];
            acc = fmaf(dw[tap], s, acc);
        }
        out[(size_t)(b * COUT + c) * HWD + hw] = acc;
    }
}

// ---------------------------------------------------------------------------
// Launcher. Workspace layout (bytes):
//   [0)          pw_bf16   : 256*256*2      =   131072
//   [131072)     offw_bf16 : 32*256*2       =    16384  (zero-padded)
//   [147456)     xT  bf16  : 2*9216*256*2   =  9437184  (x transposed)
//   [9584640)    xp  fp32  : 2*256*9216*4   = 18874368
//   [28459008)   xpT bf16  : 2*9216*256*2   =  9437184  (xp transposed)
//   [37896192)   off fp32  : 2*18*9216*4    =  1327104
//   total ~39.3 MB
// ---------------------------------------------------------------------------
extern "C" void kernel_launch(void* const* d_in, const int* in_sizes, int n_in,
                              void* d_out, int out_size, void* d_ws, size_t ws_size,
                              hipStream_t stream) {
    const float* x     = (const float*)d_in[0];
    const float* pw_w  = (const float*)d_in[1];
    const float* off_w = (const float*)d_in[2];
    const float* off_b = (const float*)d_in[3];
    const float* dw_w  = (const float*)d_in[4];
    float* out = (float*)d_out;

    char* ws = (char*)d_ws;
    bf16_t* pwbf = (bf16_t*)(ws);
    bf16_t* owbf = (bf16_t*)(ws + 131072);
    bf16_t* xT   = (bf16_t*)(ws + 147456);
    float*  xp   = (float*)(ws + 9584640);
    bf16_t* xpT  = (bf16_t*)(ws + 28459008);
    float*  off  = (float*)(ws + 37896192);

    convert_weights_kernel<<<256, 256, 0, stream>>>(pw_w, off_w, pwbf, owbf);
    transpose_x_kernel<<<dim3(HWD / 32, CIN / 32, BATCH), 256, 0, stream>>>(x, xT);
    gemm_xp_kernel<<<dim3(HWD / 32, COUT / 32, BATCH), 128, 0, stream>>>(xT, pwbf, xp, xpT);
    gemm_off_kernel<<<dim3(HWD / 32, 1, BATCH), 128, 0, stream>>>(xpT, owbf, off_b, off);
    deform_sample_kernel<<<dim3(HWD / 64, COUT / 64, BATCH), 256, 0, stream>>>(xp, off, dw_w, out);
}
